// GaussianAttention_34059090657617
// MI455X (gfx1250) — matrix operand order repaired and checked by reading, not compiled, not run on hardware
//
#include <hip/hip_runtime.h>

typedef __attribute__((ext_vector_type(16))) _Float16 v16h;
typedef __attribute__((ext_vector_type(8)))  float    v8f;

#define S_DIM 512
#define B_DIM 32
#define T_DIM 1024
#define E_DIM 512
#define Q_DIM 128

// exp(-x) == 0.0f in float32 for x > ~104 (denormal underflow ~103.98).
// Skip threshold 90: skipped contributions are < 1e-36 * |emb|, i.e. exactly
// what the f32 reference also flushes to (or denormals far below tolerance).
#define SKIP_THRESH 90.0f

// ---------------------------------------------------------------------------
// Kernel 1: query projection -> exp -> cumsum -> mean/std
// One block per batch b (32 blocks), 512 threads = one per sequence index s.
// ---------------------------------------------------------------------------
__global__ __launch_bounds__(512) void ga_stats(const float* __restrict__ query,
                                                const float* __restrict__ position,
                                                const float* __restrict__ W,
                                                const float* __restrict__ bias,
                                                float* __restrict__ meanOut,  // d_out + S*B*E, [S,B]
                                                float* __restrict__ meanWs,   // scratch [S,B]
                                                float* __restrict__ stdWs)    // scratch [S,B]
{
    __shared__ float sW[2 * Q_DIM];
    __shared__ float sScan[S_DIM];

    const int s = threadIdx.x;
    const int b = blockIdx.x;

    if (s < 2 * Q_DIM) sW[s] = W[s];
    __syncthreads();

    const float* q = query + ((size_t)s * B_DIM + b) * Q_DIM;
    float d0 = bias[0];
    float d1 = bias[1];
#pragma unroll 4
    for (int i = 0; i < Q_DIM; ++i) {
        float qv = q[i];
        d0 += qv * sW[i];
        d1 += qv * sW[Q_DIM + i];
    }
    float mean_raw = __expf(d0);
    float stdv     = __expf(d1);

    // inclusive scan over s (Hillis-Steele in LDS)
    sScan[s] = mean_raw;
    __syncthreads();
    for (int off = 1; off < S_DIM; off <<= 1) {
        float v = (s >= off) ? sScan[s - off] : 0.0f;
        __syncthreads();
        sScan[s] += v;
        __syncthreads();
    }

    const size_t idx = (size_t)s * B_DIM + b;
    float mean = position[idx] + sScan[s] * (1.0f / 20.0f);
    meanOut[idx] = mean;   // second output of the reference
    meanWs[idx]  = mean;
    stdWs[idx]   = stdv;
}

// ---------------------------------------------------------------------------
// Kernel 2: new_context[s,b,e] = sum_t exp(-std*(mean-t)^2) * emb[t,b,e]*mask
// Grid (S/128, E/128, B). 256 threads = 8 waves; wave w owns s-tile
// s_blk+16*w and all 128 e-columns of this block (8 WMMA accumulators).
// A (weights) generated analytically in registers in WMMA A-fragment layout;
// B (embeddings) staged f32->f16 through LDS in B-fragment layout.
// K-chunks whose entire weight tile underflows to 0.0f are skipped:
//   - block-uniform skip: no staging, no barriers
//   - wave-uniform skip:  no A-gen / ds_load / wmma (barriers still hit)
// ---------------------------------------------------------------------------
__global__ __launch_bounds__(256) void ga_gemm(const float* __restrict__ emb,
                                               const float* __restrict__ mask,
                                               const float* __restrict__ meanWs,
                                               const float* __restrict__ stdWs,
                                               float* __restrict__ out)
{
    // LDS B chunk: [et(8)][kbase(2)][n(16)][j(16)] halves = 8 KB.
    __shared__ alignas(32) _Float16 ldsB[8 * 2 * 16 * 16];
    __shared__ float redMn[8], redMx[8], redSm[8];

    const int tid  = threadIdx.x;
    const int wave = tid >> 5;
    const int lane = tid & 31;
    const int m    = lane & 15;   // row (A) / column (B) index inside tile
    const int kb   = lane >> 4;   // K half selector

    const int s_tile  = blockIdx.x * 128 + wave * 16;
    const int e_block = blockIdx.y * 128;
    const int b       = blockIdx.z;

    const float mean_m = meanWs[(size_t)(s_tile + m) * B_DIM + b];
    const float std_m  = stdWs [(size_t)(s_tile + m) * B_DIM + b];

    // ---- wave-level window bounds (uniform after shfl reduction) ----
    float wmn = mean_m, wmx = mean_m, wsm = std_m;
#pragma unroll
    for (int off = 16; off >= 1; off >>= 1) {
        wmn = fminf(wmn, __shfl_xor(wmn, off, 32));
        wmx = fmaxf(wmx, __shfl_xor(wmx, off, 32));
        wsm = fminf(wsm, __shfl_xor(wsm, off, 32));
    }
    if (lane == 0) { redMn[wave] = wmn; redMx[wave] = wmx; redSm[wave] = wsm; }
    __syncthreads();
    float bmn = redMn[0], bmx = redMx[0], bsm = redSm[0];
#pragma unroll
    for (int w = 1; w < 8; ++w) {
        bmn = fminf(bmn, redMn[w]);
        bmx = fmaxf(bmx, redMx[w]);
        bsm = fminf(bsm, redSm[w]);
    }

    // staging assignment: thread -> (e_local, K half)
    const int e_local = tid & 127;
    const int kbs     = tid >> 7;               // 0 or 1 -> K rows [0,16) or [16,32)
    const int wbase   = (((e_local >> 4) * 2 + kbs) * 16 + (e_local & 15)) * 16;

    v8f acc[8];
#pragma unroll
    for (int i = 0; i < 8; ++i)
        acc[i] = (v8f){0.f, 0.f, 0.f, 0.f, 0.f, 0.f, 0.f, 0.f};

    const float* src0 = emb + (size_t)b * E_DIM + e_block + e_local;

    for (int t0 = 0; t0 < T_DIM; t0 += 32) {
        const float tlo = (float)t0;
        const float thi = (float)(t0 + 31);

        // block-uniform skip: entire 128x32 weight tile underflows to 0.0f
        float db = fmaxf(fmaxf(bmn - thi, tlo - bmx), 0.0f);
        if (bsm * db * db > SKIP_THRESH) continue;

        // ---- stage B chunk: emb[t0..t0+32, b, e_block..e_block+128] ----
        union { v16h v; _Float16 h[16]; } stg;
        const int trow = t0 + kbs * 16;
#pragma unroll
        for (int j = 0; j < 16; ++j) {
            float v  = src0[(size_t)(trow + j) * (B_DIM * E_DIM)];
            float mk = mask[(size_t)(trow + j) * B_DIM + b];
            stg.h[j] = (_Float16)(v * mk);
        }
        __syncthreads();                         // prior LDS reads complete
        *(v16h*)&ldsB[wbase] = stg.v;
        __syncthreads();                         // chunk visible to all waves

        // wave-uniform skip: this wave's 16x32 weight tile underflows
        float dw = fmaxf(fmaxf(wmn - thi, tlo - wmx), 0.0f);
        if (wsm * dw * dw > SKIP_THRESH) continue;

        // ---- A fragment: Gaussian weights, generated in registers ----
        // A layout (contiguous, matches LDS B): lane m=lane&15, K = kb*16 + j
        union { v16h v; _Float16 h[16]; } af;
#pragma unroll
        for (int j = 0; j < 16; ++j) {
            int   K = kb * 16 + j;
            float d = mean_m - (float)(t0 + K);
            af.h[j] = (_Float16)__expf(-std_m * d * d);
        }

        // ---- 8 WMMAs: one A fragment reused across 8 e-subtiles ----
#pragma unroll
        for (int et = 0; et < 8; ++et) {
            v16h bf = *(const v16h*)&ldsB[((et * 2 + kb) * 16 + m) * 16];
            acc[et] = __builtin_amdgcn_wmma_f32_16x16x32_f16(
                false, af.v, false, bf, (short)0, acc[et], false, false);
        }
    }

    // ---- epilogue: C layout lane -> (M = r + kb*8, N = lane&15) ----
#pragma unroll
    for (int et = 0; et < 8; ++et) {
#pragma unroll
        for (int r = 0; r < 8; ++r) {
            int mm = r + kb * 8;
            out[((size_t)(s_tile + mm) * B_DIM + b) * E_DIM + e_block + et * 16 + m] =
                acc[et][r];
        }
    }
}

// ---------------------------------------------------------------------------
extern "C" void kernel_launch(void* const* d_in, const int* in_sizes, int n_in,
                              void* d_out, int out_size, void* d_ws, size_t ws_size,
                              hipStream_t stream)
{
    const float* query = (const float*)d_in[0];  // [S,B,Q]
    const float* emb   = (const float*)d_in[1];  // [T,B,E]
    const float* mask  = (const float*)d_in[2];  // [T,B]
    const float* pos   = (const float*)d_in[3];  // [S,B]
    const float* W     = (const float*)d_in[4];  // [2,Q]
    const float* bias  = (const float*)d_in[5];  // [2]

    float* out     = (float*)d_out;                       // new_context [S,B,E]
    float* meanOut = out + (size_t)S_DIM * B_DIM * E_DIM; // mean [S,B]

    float* meanWs = (float*)d_ws;                         // [S,B]
    float* stdWs  = meanWs + (size_t)S_DIM * B_DIM;       // [S,B]

    ga_stats<<<dim3(B_DIM), dim3(S_DIM), 0, stream>>>(
        query, pos, W, bias, meanOut, meanWs, stdWs);

    ga_gemm<<<dim3(S_DIM / 128, E_DIM / 128, B_DIM), dim3(256), 0, stream>>>(
        emb, mask, meanWs, stdWs, out);
}
